// AttentionLayer_39513699123575
// MI455X (gfx1250) — matrix operand order repaired and checked by reading, not compiled, not run on hardware
//
#include <hip/hip_runtime.h>
#include <hip/hip_bf16.h>
#include <math.h>

#define BATCH  4
#define SEQ    2048
#define DMODEL 1024
#define NHEAD  16
#define DHEAD  64
#define NOUT   (NHEAD * DHEAD)   /* 1024 */

typedef _Float16 v16h __attribute__((ext_vector_type(16)));
typedef _Float16 v8h  __attribute__((ext_vector_type(8)));
typedef _Float16 v4h  __attribute__((ext_vector_type(4)));
typedef float    v8f  __attribute__((ext_vector_type(8)));
typedef float    v4f_ __attribute__((ext_vector_type(4)));
typedef int      v4i  __attribute__((ext_vector_type(4)));

// ---- optional CDNA5 async global->LDS path (compile-safe probe) -----------
#if defined(__has_builtin)
#if __has_builtin(__builtin_amdgcn_global_load_async_to_lds_b128) && \
    __has_builtin(__builtin_amdgcn_s_wait_asynccnt)
#define USE_ASYNC_LDS 1
#endif
#endif
#ifndef USE_ASYNC_LDS
#define USE_ASYNC_LDS 0
#endif

#if USE_ASYNC_LDS
// param0 is 'int4 __device__ *' (addrspace(1)); param1 LDS side (addrspace(3))
typedef __attribute__((address_space(1))) v4i gv4i;   // global int4*
typedef __attribute__((address_space(3))) v4i lv4i;   // LDS int4*
#endif

static __device__ __forceinline__ v16h cat16(v8h lo, v8h hi) {
  return __builtin_shufflevector(lo, hi, 0,1,2,3,4,5,6,7,8,9,10,11,12,13,14,15);
}

// A-frag (16x32 f16). rowptr -> element [m = lane%16][k = 0] of the tile
// (K contiguous). Lanes 0-15 hold K 0-15; lanes 16-31 hold K 16-31.
static __device__ __forceinline__ v16h load_a_frag(const _Float16* rowptr, int grp) {
  v8h lo = *(const v8h*)(rowptr + grp * 16);
  v8h hi = *(const v8h*)(rowptr + grp * 16 + 8);
  return cat16(lo, hi);
}

// B-frag (32x16 f16) from an n-major buffer. ptr -> element [n = lane%16][k = grp*16],
// 16 contiguous K halfs per lane.
static __device__ __forceinline__ v16h load_b_frag(const _Float16* ptr) {
  v8h lo = *(const v8h*)(ptr);
  v8h hi = *(const v8h*)(ptr + 8);
  return cat16(lo, hi);
}

static __device__ __forceinline__ v8f wmma_f16(v16h a, v16h b, v8f c) {
  return __builtin_amdgcn_wmma_f32_16x16x32_f16(false, a, false, b, (short)0, c,
                                                false, false);
}

// ---------------------------------------------------------------------------
// QKV projection GEMM: out[b,h,s,d] (f16) = x[8192,1024] @ w[1024,1024] + bias
// Block = 128 threads (4 waves), tile 64(M) x 64(N), K-step 32.
// Staging: batch-issue 4 b128 global loads, then convert f32->f16 and store.
// ---------------------------------------------------------------------------
__global__ __launch_bounds__(128)
void qkv_gemm_wmma(const float* __restrict__ x, const float* __restrict__ w,
                   const float* __restrict__ bias, _Float16* __restrict__ out) {
  __shared__ _Float16 As[64][40];   // row-major, K contiguous (80B rows, 16B aligned)
  __shared__ _Float16 Bt[64][40];   // n-major (transposed), K contiguous

  const int tid  = threadIdx.x;
  const int lane = tid & 31;
  const int wave = tid >> 5;
  const int n16  = lane & 15;
  const int grp  = lane >> 4;

  const int mbase = blockIdx.y * 64;
  const int nbase = blockIdx.x * 64;
  const int wm = (wave >> 1) * 32;
  const int wn = (wave & 1) * 32;

  v8f acc[2][2];
#pragma unroll
  for (int i = 0; i < 2; ++i)
#pragma unroll
    for (int j = 0; j < 2; ++j) acc[i][j] = (v8f){0.f,0.f,0.f,0.f,0.f,0.f,0.f,0.f};

  for (int kt = 0; kt < DMODEL; kt += 32) {
    __syncthreads();
    // ---- load phase: 8 independent b128 loads in flight -------------------
    v4f_ xa[4], wv[4];
#pragma unroll
    for (int it = 0; it < 4; ++it) {
      int i = it * 128 + tid;
      int r = i >> 3;                                 // A: 8 float4 per row of 32
      int c = (i & 7) * 4;
      xa[it] = *(const v4f_*)&x[(size_t)(mbase + r) * DMODEL + kt + c];
      int k = i >> 4;                                 // B: 16 float4 per row of 64
      int n = (i & 15) * 4;
      wv[it] = *(const v4f_*)&w[(size_t)(kt + k) * NOUT + nbase + n];
    }
    // ---- convert + store phase -------------------------------------------
#pragma unroll
    for (int it = 0; it < 4; ++it) {
      int i = it * 128 + tid;
      int r = i >> 3;
      int c = (i & 7) * 4;
      *(v4h*)&As[r][c] = __builtin_convertvector(xa[it], v4h);
      int k = i >> 4;
      int n = (i & 15) * 4;
      v4h hv = __builtin_convertvector(wv[it], v4h);
      Bt[n + 0][k] = hv[0];
      Bt[n + 1][k] = hv[1];
      Bt[n + 2][k] = hv[2];
      Bt[n + 3][k] = hv[3];
    }
    if (kt + 32 < DMODEL)   // hint L2 for the next K-slab of x
      __builtin_prefetch(&x[(size_t)(mbase + (tid >> 1)) * DMODEL + kt + 32], 0, 1);
    __syncthreads();

#pragma unroll
    for (int mt = 0; mt < 2; ++mt) {
      v16h a = load_a_frag(&As[wm + mt * 16 + n16][0], grp);
#pragma unroll
      for (int nt = 0; nt < 2; ++nt) {
        v16h bf = load_b_frag(&Bt[wn + nt * 16 + n16][grp * 16]);
        acc[mt][nt] = wmma_f16(a, bf, acc[mt][nt]);
      }
    }
  }

  // write out as [B, H, S, DH] f16 with bias add
#pragma unroll
  for (int mt = 0; mt < 2; ++mt)
#pragma unroll
    for (int nt = 0; nt < 2; ++nt)
#pragma unroll
      for (int r = 0; r < 8; ++r) {
        int m = mbase + wm + mt * 16 + r + grp * 8;     // global row = b*S+s
        int n = nbase + wn + nt * 16 + n16;             // global col = h*DH+d
        int b = m >> 11, s = m & (SEQ - 1);
        int h = n >> 6,  d = n & (DHEAD - 1);
        float v = acc[mt][nt][r] + bias[n];
        out[(((size_t)(b * NHEAD + h)) * SEQ + s) * DHEAD + d] = (_Float16)v;
      }
}

// ---------------------------------------------------------------------------
// Flash attention: per (b,h), online-softmax over 64-key blocks.
// Block = 128 threads (4 waves); each wave owns 16 query rows.
// K tile staged with async global->LDS copies; V loads batch-issued.
// ---------------------------------------------------------------------------
__global__ __launch_bounds__(128)
void flash_attn_wmma(const _Float16* __restrict__ Qh, const _Float16* __restrict__ Kh,
                     const _Float16* __restrict__ Vh, const int* __restrict__ mask,
                     float* __restrict__ out) {
  __shared__ _Float16 Ks[64][72];      // [key][d]  (d contiguous) -> B-frag for Q*K^T
  __shared__ _Float16 Vt[64][72];      // [d][key]  (key contiguous) -> B-frag for P*V
  __shared__ _Float16 Ps[4][16][72];   // per-wave P tile, [qrow][key]
  __shared__ float    madd[64];

  const int tid  = threadIdx.x;
  const int lane = tid & 31;
  const int wave = tid >> 5;
  const int n16  = lane & 15;
  const int grp  = lane >> 4;

  const int bh = blockIdx.y;            // 0..63
  const int b  = bh >> 4;
  const int h  = bh & 15;
  const int qbase = blockIdx.x * 64 + wave * 16;

  const _Float16* qptr = Qh + (size_t)bh * SEQ * DHEAD;
  const _Float16* kptr = Kh + (size_t)bh * SEQ * DHEAD;
  const _Float16* vptr = Vh + (size_t)bh * SEQ * DHEAD;

  // preload this wave's Q rows as A-frags: [kk] covers DH halves 0-31 / 32-63
  v16h qf[2];
#pragma unroll
  for (int kk = 0; kk < 2; ++kk)
    qf[kk] = load_a_frag(qptr + (size_t)(qbase + n16) * DHEAD + kk * 32, grp);

  v8f acc[4];
#pragma unroll
  for (int nt = 0; nt < 4; ++nt) acc[nt] = (v8f){0.f,0.f,0.f,0.f,0.f,0.f,0.f,0.f};
  float Mrow[8], Lrow[8];
#pragma unroll
  for (int r = 0; r < 8; ++r) { Mrow[r] = -1e30f; Lrow[r] = 0.f; }

  for (int kb = 0; kb < SEQ; kb += 64) {
    __syncthreads();
    // ---- load phase: async K copies + 4 batched V b128 loads --------------
    v8h vv[4];
#pragma unroll
    for (int it = 0; it < 4; ++it) {
      int i = it * 128 + tid;
      int ky = i >> 3;
      int dd = (i & 7) * 8;
      const _Float16* kg = kptr + (size_t)(kb + ky) * DHEAD + dd;
#if USE_ASYNC_LDS
      __builtin_amdgcn_global_load_async_to_lds_b128(
          (gv4i*)kg, (lv4i*)&Ks[ky][dd], 0, 0);
#else
      *(v8h*)&Ks[ky][dd] = *(const v8h*)kg;
#endif
      vv[it] = *(const v8h*)(vptr + (size_t)(kb + ky) * DHEAD + dd);
    }
    // ---- scatter phase: V transpose into LDS ------------------------------
#pragma unroll
    for (int it = 0; it < 4; ++it) {
      int i = it * 128 + tid;
      int ky = i >> 3;
      int dd = (i & 7) * 8;
#pragma unroll
      for (int j = 0; j < 8; ++j) Vt[dd + j][ky] = vv[it][j];
    }
    if (tid < 64)
      madd[tid] = (1.0f - (float)mask[b * SEQ + kb + tid]) * -10000.0f;
#if USE_ASYNC_LDS
    __builtin_amdgcn_s_wait_asynccnt(0);
#endif
    __syncthreads();

    // ---- scores: S = Q K^T * (1/sqrt(64)) + mask  (2 WMMAs per 16-key tile)
    v8f sc[4];
#pragma unroll
    for (int nt = 0; nt < 4; ++nt) {
      v8f c = (v8f){0.f,0.f,0.f,0.f,0.f,0.f,0.f,0.f};
#pragma unroll
      for (int kk = 0; kk < 2; ++kk) {
        v16h bf = load_b_frag(&Ks[nt * 16 + n16][kk * 32 + grp * 16]);
        c = wmma_f16(qf[kk], bf, c);
      }
      sc[nt] = c;
    }

    float ma[4];
#pragma unroll
    for (int nt = 0; nt < 4; ++nt) ma[nt] = madd[nt * 16 + n16];

    // ---- online softmax (C layout: row = r + grp*8, col = nt*16 + lane%16)
#pragma unroll
    for (int r = 0; r < 8; ++r) {
      float mx = -1e30f;
#pragma unroll
      for (int nt = 0; nt < 4; ++nt) {
        float sv = sc[nt][r] * 0.125f + ma[nt];
        sc[nt][r] = sv;
        mx = fmaxf(mx, sv);
      }
#pragma unroll
      for (int off = 8; off > 0; off >>= 1)
        mx = fmaxf(mx, __shfl_xor(mx, off, 32));      // reduce over 16-lane half-group
      float nm   = fmaxf(Mrow[r], mx);
      float corr = __expf(Mrow[r] - nm);
      Mrow[r] = nm;
      float psum = 0.f;
#pragma unroll
      for (int nt = 0; nt < 4; ++nt) {
        float p = __expf(sc[nt][r] - nm);
        psum += p;
        acc[nt][r] *= corr;
        Ps[wave][r + grp * 8][nt * 16 + n16] = (_Float16)p;   // C-layout -> LDS
      }
#pragma unroll
      for (int off = 8; off > 0; off >>= 1)
        psum += __shfl_xor(psum, off, 32);
      Lrow[r] = Lrow[r] * corr + psum;
    }

    // ---- P*V (same-wave LDS round trip turns C layout into A-frags)
    v16h pa0 = load_a_frag(&Ps[wave][n16][0],  grp);
    v16h pa1 = load_a_frag(&Ps[wave][n16][32], grp);
#pragma unroll
    for (int nt = 0; nt < 4; ++nt) {
      v16h vb0 = load_b_frag(&Vt[nt * 16 + n16][grp * 16]);
      v16h vb1 = load_b_frag(&Vt[nt * 16 + n16][32 + grp * 16]);
      acc[nt] = wmma_f16(pa0, vb0, acc[nt]);
      acc[nt] = wmma_f16(pa1, vb1, acc[nt]);
    }
  }

  // ---- epilogue: normalize and write [B, S, H*DH] fp32 (lane-coalesced)
#pragma unroll
  for (int nt = 0; nt < 4; ++nt)
#pragma unroll
    for (int r = 0; r < 8; ++r) {
      int srow = qbase + r + grp * 8;
      int dcol = nt * 16 + n16;
      out[(size_t)(b * SEQ + srow) * NOUT + h * DHEAD + dcol] = acc[nt][r] / Lrow[r];
    }
}

// ---------------------------------------------------------------------------
extern "C" void kernel_launch(void* const* d_in, const int* in_sizes, int n_in,
                              void* d_out, int out_size, void* d_ws, size_t ws_size,
                              hipStream_t stream) {
  (void)in_sizes; (void)n_in; (void)out_size; (void)ws_size;
  const float* x    = (const float*)d_in[0];
  const int*   mask = (const int*)d_in[1];
  const float* wq   = (const float*)d_in[2];
  const float* bq   = (const float*)d_in[3];
  const float* wk   = (const float*)d_in[4];
  const float* bk   = (const float*)d_in[5];
  const float* wv   = (const float*)d_in[6];
  const float* bv   = (const float*)d_in[7];
  float* out = (float*)d_out;

  const size_t perTensor = (size_t)BATCH * NHEAD * SEQ * DHEAD;  // 8M halfs = 16 MB
  _Float16* Qh = (_Float16*)d_ws;
  _Float16* Kh = Qh + perTensor;
  _Float16* Vh = Kh + perTensor;

  dim3 gemmGrid(NOUT / 64, (BATCH * SEQ) / 64);   // (16, 128)
  qkv_gemm_wmma<<<gemmGrid, 128, 0, stream>>>(x, wq, bq, Qh);
  qkv_gemm_wmma<<<gemmGrid, 128, 0, stream>>>(x, wk, bk, Kh);
  qkv_gemm_wmma<<<gemmGrid, 128, 0, stream>>>(x, wv, bv, Vh);

  dim3 attnGrid(SEQ / 64, BATCH * NHEAD);         // (32, 64)
  flash_attn_wmma<<<attnGrid, 128, 0, stream>>>(Qh, Kh, Vh, mask, out);
}
